// GraphCriticNet_35682588295233
// MI455X (gfx1250) — matrix operand order repaired and checked by reading, not compiled, not run on hardware
//
#include <hip/hip_runtime.h>
#include <stdint.h>

#define H 256
#define LN_EPS 1e-5f
#define LDSP 264   // LDS A-tile pitch in ushorts (256 + 8 pad -> bank-conflict-free-ish)

typedef __attribute__((ext_vector_type(16))) __bf16 bf16x16;
typedef __attribute__((ext_vector_type(8)))  float  v8f;
typedef __attribute__((__vector_size__(16))) int    i32x4;

#define AS1 __attribute__((address_space(1)))
#define AS3 __attribute__((address_space(3)))

#ifndef HAVE_ASYNC_LDS
#if defined(__has_builtin)
#if __has_builtin(__builtin_amdgcn_global_load_async_to_lds_b128)
#define HAVE_ASYNC_LDS 1
#endif
#endif
#endif
#ifndef HAVE_ASYNC_LDS
#define HAVE_ASYNC_LDS 0
#endif

union Frag {
    bf16x16 v;
    uint4   u[2];
};

__device__ __forceinline__ unsigned short f2bf(float f) {
    unsigned int u = __float_as_uint(f);
    unsigned int r = u + 0x7FFFu + ((u >> 16) & 1u);   // round-to-nearest-even
    return (unsigned short)(r >> 16);
}

// Stage one 16x256 bf16 A-tile (8KB, contiguous rows of 512B) from global into a
// padded LDS tile (pitch LDSP ushorts). 256 threads, 32B each.
__device__ __forceinline__ void stage_a_tile(const unsigned short* __restrict__ gsrc,
                                             unsigned short* sdst, int tid) {
    const int r  = tid >> 4;            // row 0..15
    const int cb = (tid & 15) * 32;     // byte offset within row, 0..480
    const long long goff = (long long)r * 512 + cb;
    const int       loff = r * (LDSP * 2) + cb;
#if HAVE_ASYNC_LDS
    const AS1 char* g = (const AS1 char*)gsrc;
    AS3 char*       s = (AS3 char*)sdst;
    __builtin_amdgcn_global_load_async_to_lds_b128(
        (AS1 i32x4*)(g + goff),      (AS3 i32x4*)(s + loff),      0, 0);
    __builtin_amdgcn_global_load_async_to_lds_b128(
        (AS1 i32x4*)(g + goff + 16), (AS3 i32x4*)(s + loff + 16), 0, 0);
#if __has_builtin(__builtin_amdgcn_s_wait_asynccnt)
    __builtin_amdgcn_s_wait_asynccnt(0);
#else
    asm volatile("s_wait_asynccnt 0" ::: "memory");
#endif
#else
    uint4 t0 = *(const uint4*)((const char*)gsrc + goff);
    uint4 t1 = *(const uint4*)((const char*)gsrc + goff + 16);
    *(uint4*)((char*)sdst + loff)      = t0;
    *(uint4*)((char*)sdst + loff + 16) = t1;
#endif
    __syncthreads();
}

// ---------------------------------------------------------------------------
// Pack a row-major [256,256] fp32 weight into per-fragment bf16 layout:
// Wp[((kk*16+nt)*32 + lane)*16 + e] = W[(kk*32 + (lane>>4)*16 + e)*256 + nt*16 + (lane&15)]
// ---------------------------------------------------------------------------
__global__ void pack_w_kernel(const float* __restrict__ W, unsigned short* __restrict__ Wp) {
    int t      = blockIdx.x * 256 + threadIdx.x;      // 0..65535
    int within = t & 511;
    int frag   = t >> 9;
    int lane   = within >> 4;
    int e      = within & 15;
    int kk     = frag >> 4;
    int nt     = frag & 15;
    int krow   = kk * 32 + (lane >> 4) * 16 + e;
    int col    = nt * 16 + (lane & 15);
    Wp[t] = f2bf(W[krow * H + col]);
}

// ---------------------------------------------------------------------------
// Node projection: h = relu(concat(x, a) @ np_w + np_b)   [N,4]x[4,256]
// ---------------------------------------------------------------------------
__global__ __launch_bounds__(256)
void node_proj_kernel(const float* __restrict__ x, const int* __restrict__ a,
                      const float* __restrict__ w, const float* __restrict__ b,
                      float* __restrict__ h) {
    long long row = blockIdx.x;
    int c = threadIdx.x;
    float acc = b[c];
    acc += x[row * 3 + 0] * w[0 * H + c];
    acc += x[row * 3 + 1] * w[1 * H + c];
    acc += x[row * 3 + 2] * w[2 * H + c];
    acc += (float)a[row]  * w[3 * H + c];
    h[row * H + c] = fmaxf(acc, 0.f);
}

// ---------------------------------------------------------------------------
// Edge MLP first linear: t = relu(edge_attr @ em_w1 + em_b1)  [E,4]x[4,256] -> bf16
// ---------------------------------------------------------------------------
__global__ __launch_bounds__(256)
void edge_lin1_kernel(const float* __restrict__ ea, const float* __restrict__ w,
                      const float* __restrict__ b, unsigned short* __restrict__ out) {
    long long row = blockIdx.x;
    int c = threadIdx.x;
    float acc = b[c];
    #pragma unroll
    for (int k = 0; k < 4; ++k) acc += ea[row * 4 + k] * w[k * H + c];
    out[row * H + c] = f2bf(fmaxf(acc, 0.f));
}

// ---------------------------------------------------------------------------
// WMMA GEMM: C[M,256] = A_bf16[M,256] @ Wp + bias ; optional relu; out f32 or bf16.
// Block = 256 threads = 8 waves; wave w -> col tiles 2w, 2w+1; blockIdx.x = row tile.
// A tile staged in LDS (async-to-LDS when available); in-place (out == A) is safe
// because all global reads of A happen before the staging barrier.
// ---------------------------------------------------------------------------
template <bool RELU, bool OUT_BF16>
__global__ __launch_bounds__(256)
void gemm_h256_kernel(const unsigned short* __restrict__ A,
                      const unsigned short* __restrict__ Wp,
                      const float* __restrict__ bias,
                      float* __restrict__ outF,
                      unsigned short* __restrict__ outB) {
    __shared__ __align__(16) unsigned short Atile[16 * LDSP];
    const int tid  = threadIdx.x;
    const int wave = tid >> 5;
    const int lane = tid & 31;
    const int hh   = lane >> 4;
    const int ln   = lane & 15;
    const long long row0 = (long long)blockIdx.x * 16;
    const int nt0 = wave * 2;

    stage_a_tile(A + row0 * H, Atile, tid);

    v8f c0 = {};
    v8f c1 = {};
    const unsigned short* arow = Atile + ln * LDSP;

    #pragma unroll
    for (int kk = 0; kk < 8; ++kk) {
        Frag af, b0, b1;
        const unsigned short* ap = arow + kk * 32 + hh * 16;
        af.u[0] = *(const uint4*)(ap);
        af.u[1] = *(const uint4*)(ap + 8);
        const unsigned short* bp0 = Wp + ((kk * 16 + nt0) * 32 + lane) * 16;
        b0.u[0] = *(const uint4*)(bp0);
        b0.u[1] = *(const uint4*)(bp0 + 8);
        const unsigned short* bp1 = bp0 + 512;
        b1.u[0] = *(const uint4*)(bp1);
        b1.u[1] = *(const uint4*)(bp1 + 8);
        c0 = __builtin_amdgcn_wmma_f32_16x16x32_bf16(false, af.v, false, b0.v, (short)0, c0, false, false);
        c1 = __builtin_amdgcn_wmma_f32_16x16x32_bf16(false, af.v, false, b1.v, (short)0, c1, false, false);
    }
    #pragma unroll
    for (int t = 0; t < 2; ++t) {
        v8f c = t ? c1 : c0;
        int col = (nt0 + t) * 16 + ln;
        float bv = bias[col];
        #pragma unroll
        for (int v = 0; v < 8; ++v) {
            long long row = row0 + v + 8 * hh;
            float val = c[v] + bv;
            if (RELU) val = fmaxf(val, 0.f);
            if (OUT_BF16) outB[row * H + col] = f2bf(val);
            else          outF[row * H + col] = val;
        }
    }
}

// ---------------------------------------------------------------------------
// Fused GINEConv message pass for one layer:
//   e_l  = e @ lin_w[l] + lin_b[l]          (WMMA, bf16 in / f32 acc)
//   msg  = relu(h[src] + e_l)               (gather in epilogue)
//   agg[dst] += msg                         (atomic scatter in epilogue)
// ---------------------------------------------------------------------------
__global__ __launch_bounds__(256)
void edge_layer_scatter_kernel(const unsigned short* __restrict__ Ebf,
                               const unsigned short* __restrict__ Wp,
                               const float* __restrict__ bias,
                               const float* __restrict__ Hf,
                               const long long* __restrict__ eidx, long long E,
                               float* __restrict__ agg) {
    __shared__ __align__(16) unsigned short Atile[16 * LDSP];
    const int tid  = threadIdx.x;
    const int wave = tid >> 5;
    const int lane = tid & 31;
    const int hh   = lane >> 4;
    const int ln   = lane & 15;
    const long long row0 = (long long)blockIdx.x * 16;
    const int nt0 = wave * 2;

    stage_a_tile(Ebf + row0 * H, Atile, tid);

    v8f c0 = {};
    v8f c1 = {};
    const unsigned short* arow = Atile + ln * LDSP;

    #pragma unroll
    for (int kk = 0; kk < 8; ++kk) {
        Frag af, b0, b1;
        const unsigned short* ap = arow + kk * 32 + hh * 16;
        af.u[0] = *(const uint4*)(ap);
        af.u[1] = *(const uint4*)(ap + 8);
        const unsigned short* bp0 = Wp + ((kk * 16 + nt0) * 32 + lane) * 16;
        b0.u[0] = *(const uint4*)(bp0);
        b0.u[1] = *(const uint4*)(bp0 + 8);
        const unsigned short* bp1 = bp0 + 512;
        b1.u[0] = *(const uint4*)(bp1);
        b1.u[1] = *(const uint4*)(bp1 + 8);
        c0 = __builtin_amdgcn_wmma_f32_16x16x32_bf16(false, af.v, false, b0.v, (short)0, c0, false, false);
        c1 = __builtin_amdgcn_wmma_f32_16x16x32_bf16(false, af.v, false, b1.v, (short)0, c1, false, false);
    }
    #pragma unroll
    for (int t = 0; t < 2; ++t) {
        v8f c = t ? c1 : c0;
        int col = (nt0 + t) * 16 + ln;
        float bv = bias[col];
        #pragma unroll
        for (int v = 0; v < 8; ++v) {
            long long edge = row0 + v + 8 * hh;
            long long s = eidx[edge];
            long long d = eidx[E + edge];
            float msg = fmaxf(Hf[s * H + col] + c[v] + bv, 0.f);
            atomicAdd(agg + d * H + col, msg);
        }
    }
}

// ---------------------------------------------------------------------------
// zb = bf16(h + agg)
// ---------------------------------------------------------------------------
__global__ __launch_bounds__(256)
void zcombine_kernel(const float* __restrict__ h, const float* __restrict__ agg,
                     unsigned short* __restrict__ zb) {
    long long i = (long long)blockIdx.x * 256 + threadIdx.x;
    zb[i] = f2bf(h[i] + agg[i]);
}

// ---------------------------------------------------------------------------
// LayerNorm over H=256 + residual relu: h += relu(LN(z2)*g + b)
// ---------------------------------------------------------------------------
__global__ __launch_bounds__(256)
void ln_residual_kernel(const float* __restrict__ z2, const float* __restrict__ g,
                        const float* __restrict__ b, float* __restrict__ h) {
    __shared__ float red[256];
    long long row = blockIdx.x;
    int c = threadIdx.x;
    float v = z2[row * H + c];
    red[c] = v;
    __syncthreads();
    #pragma unroll
    for (int s = 128; s > 0; s >>= 1) { if (c < s) red[c] += red[c + s]; __syncthreads(); }
    float mu = red[0] * (1.0f / H);
    __syncthreads();
    float dv = v - mu;
    red[c] = dv * dv;
    __syncthreads();
    #pragma unroll
    for (int s = 128; s > 0; s >>= 1) { if (c < s) red[c] += red[c + s]; __syncthreads(); }
    float var = red[0] * (1.0f / H);
    float y = dv * rsqrtf(var + LN_EPS) * g[c] + b[c];
    h[row * H + c] += fmaxf(y, 0.f);
}

// ---------------------------------------------------------------------------
// Mean pool per graph (n nodes each)
// ---------------------------------------------------------------------------
__global__ __launch_bounds__(256)
void pool_kernel(const float* __restrict__ h, float* __restrict__ g, int n) {
    int bg = blockIdx.x;
    int c = threadIdx.x;
    const float* base = h + (long long)bg * n * H + c;
    float s = 0.f;
    for (int j = 0; j < n; ++j) s += base[(long long)j * H];
    g[bg * H + c] = s / (float)n;
}

// ---------------------------------------------------------------------------
// Head: q[b] = relu(g @ hd_w1 + hd_b1) @ hd_w2 + hd_b2
// ---------------------------------------------------------------------------
__global__ __launch_bounds__(256)
void head_kernel(const float* __restrict__ g, const float* __restrict__ w1,
                 const float* __restrict__ b1, const float* __restrict__ w2,
                 const float* __restrict__ b2, float* __restrict__ q) {
    __shared__ float red[256];
    int bg = blockIdx.x;
    int c = threadIdx.x;
    const float* gr = g + bg * H;
    float acc = b1[c];
    for (int k = 0; k < H; ++k) acc += gr[k] * w1[k * H + c];
    red[c] = fmaxf(acc, 0.f) * w2[c];   // hd_w2 is [H,1]
    __syncthreads();
    #pragma unroll
    for (int s = 128; s > 0; s >>= 1) { if (c < s) red[c] += red[c + s]; __syncthreads(); }
    if (c == 0) q[bg] = red[0] + b2[0];
}

// ---------------------------------------------------------------------------
extern "C" void kernel_launch(void* const* d_in, const int* in_sizes, int n_in,
                              void* d_out, int out_size, void* d_ws, size_t ws_size,
                              hipStream_t stream) {
    (void)n_in; (void)out_size; (void)ws_size;
    const long long N = in_sizes[0] / 3;       // 32768
    const long long E = in_sizes[1] / 4;       // 524288
    const int B = 32;
    const int n = (int)(N / B);

    const float*     x      = (const float*)d_in[0];
    const float*     ea     = (const float*)d_in[1];
    const long long* eidx   = (const long long*)d_in[2];
    const int*       a      = (const int*)d_in[3];
    const float*     np_w   = (const float*)d_in[4];
    const float*     np_b   = (const float*)d_in[5];
    const float*     em_w1  = (const float*)d_in[6];
    const float*     em_b1  = (const float*)d_in[7];
    const float*     em_w2  = (const float*)d_in[8];
    const float*     em_b2  = (const float*)d_in[9];
    const float*     lin_w  = (const float*)d_in[10];
    const float*     lin_b  = (const float*)d_in[11];
    const float*     m_w1   = (const float*)d_in[12];
    const float*     m_b1   = (const float*)d_in[13];
    const float*     m_w2   = (const float*)d_in[14];
    const float*     m_b2   = (const float*)d_in[15];
    const float*     ln_g   = (const float*)d_in[16];
    const float*     ln_b   = (const float*)d_in[17];
    const float*     hd_w1  = (const float*)d_in[18];
    const float*     hd_b1  = (const float*)d_in[19];
    const float*     hd_w2  = (const float*)d_in[20];
    const float*     hd_b2  = (const float*)d_in[21];
    float* q = (float*)d_out;

    char* ws = (char*)d_ws;
    size_t off = 0;
    auto alloc = [&](size_t bytes) -> char* {
        char* p = ws + off;
        off += (bytes + 255) & ~(size_t)255;
        return p;
    };
    float*          h      = (float*)alloc(N * H * 4);
    float*          agg    = (float*)alloc(N * H * 4);
    float*          z2     = (float*)alloc(N * H * 4);
    unsigned short* zb     = (unsigned short*)alloc(N * H * 2);
    unsigned short* z1b    = (unsigned short*)alloc(N * H * 2);
    unsigned short* ebf    = (unsigned short*)alloc(E * H * 2);
    float*          gpool  = (float*)alloc(B * H * 4);
    unsigned short* wp_em2 = (unsigned short*)alloc((size_t)H * H * 2);
    unsigned short* wp_lin = (unsigned short*)alloc(3 * (size_t)H * H * 2);
    unsigned short* wp_m1  = (unsigned short*)alloc(3 * (size_t)H * H * 2);
    unsigned short* wp_m2  = (unsigned short*)alloc(3 * (size_t)H * H * 2);

    // Pack weights into WMMA B-fragment layout (bf16)
    pack_w_kernel<<<256, 256, 0, stream>>>(em_w2, wp_em2);
    for (int l = 0; l < 3; ++l) {
        pack_w_kernel<<<256, 256, 0, stream>>>(lin_w + (size_t)l * H * H, wp_lin + (size_t)l * H * H);
        pack_w_kernel<<<256, 256, 0, stream>>>(m_w1  + (size_t)l * H * H, wp_m1  + (size_t)l * H * H);
        pack_w_kernel<<<256, 256, 0, stream>>>(m_w2  + (size_t)l * H * H, wp_m2  + (size_t)l * H * H);
    }

    // h = relu([x, a] @ np_w + np_b)
    node_proj_kernel<<<(unsigned)N, 256, 0, stream>>>(x, a, np_w, np_b, h);
    // t = relu(edge_attr @ em_w1 + em_b1) -> ebf
    edge_lin1_kernel<<<(unsigned)E, 256, 0, stream>>>(ea, em_w1, em_b1, ebf);
    // e = t @ em_w2 + em_b2 (in-place, bf16 out)
    gemm_h256_kernel<false, true><<<(unsigned)(E / 16), 256, 0, stream>>>(
        ebf, wp_em2, em_b2, nullptr, ebf);

    for (int l = 0; l < 3; ++l) {
        (void)hipMemsetAsync(agg, 0, (size_t)N * H * 4, stream);
        // fused: e_l GEMM + gather h[src] + relu + atomic scatter-add into agg[dst]
        edge_layer_scatter_kernel<<<(unsigned)(E / 16), 256, 0, stream>>>(
            ebf, wp_lin + (size_t)l * H * H, lin_b + (size_t)l * H, h, eidx, E, agg);
        // zb = bf16(h + agg)
        zcombine_kernel<<<(unsigned)(N * H / 256), 256, 0, stream>>>(h, agg, zb);
        // z1 = relu(zb @ m_w1 + m_b1) -> bf16
        gemm_h256_kernel<true, true><<<(unsigned)(N / 16), 256, 0, stream>>>(
            zb, wp_m1 + (size_t)l * H * H, m_b1 + (size_t)l * H, nullptr, z1b);
        // z2 = z1 @ m_w2 + m_b2 -> f32
        gemm_h256_kernel<false, false><<<(unsigned)(N / 16), 256, 0, stream>>>(
            z1b, wp_m2 + (size_t)l * H * H, m_b2 + (size_t)l * H, z2, nullptr);
        // h += relu(LN(z2)*g + b)
        ln_residual_kernel<<<(unsigned)N, 256, 0, stream>>>(
            z2, ln_g + (size_t)l * H, ln_b + (size_t)l * H, h);
    }

    pool_kernel<<<B, 256, 0, stream>>>(h, gpool, n);
    head_kernel<<<B, 256, 0, stream>>>(gpool, hd_w1, hd_b1, hd_w2, hd_b2, q);
}
